// ASC_LSTM_3161095929897
// MI455X (gfx1250) — matrix operand order repaired and checked by reading, not compiled, not run on hardware
//
#include <hip/hip_runtime.h>
#include <hip/hip_bf16.h>

// Problem constants
#define B_   256
#define I_   256
#define H_   512
#define S_   64
#define RESN 4
#define ALPHA_ 0.5f
#define NWG_ENC 64   // persistent encoder workgroups: (B/64) * (H/32)

typedef __attribute__((ext_vector_type(16))) __bf16 v16bf;
typedef __attribute__((ext_vector_type(8)))  float  v8f;
typedef __attribute__((ext_vector_type(2)))  float  v2f32;
typedef __attribute__((ext_vector_type(2)))  __bf16 v2bf;
typedef __attribute__((ext_vector_type(16))) float  v16f;

__device__ __forceinline__ float sigf(float x) { return 1.0f / (1.0f + __expf(-x)); }

// two f32 -> one dword of two bf16; single v_cvt_pk_bf16_f32 via convertvector
__device__ __forceinline__ int pk2(float x, float y) {
  v2f32 f; f[0] = x; f[1] = y;
  union { v2bf h; int u; } t;
  t.h = __builtin_convertvector(f, v2bf);
  return t.u;
}

// 16 f32 -> v16bf in one vector fptrunc (packed cvt chain)
__device__ __forceinline__ v16bf cvt16(float4 a, float4 b, float4 c, float4 d) {
  v16f f = {a.x, a.y, a.z, a.w, b.x, b.y, b.z, b.w,
            c.x, c.y, c.z, c.w, d.x, d.y, d.z, d.w};
  return __builtin_convertvector(f, v16bf);
}

union BF16x16 { int4 q[2]; v16bf v; };

#define WMMA_BF16(A, Bm, C) \
  __builtin_amdgcn_wmma_f32_16x16x32_bf16(false, (A), false, (Bm), (short)0, (C), false, false)

// Direct-from-global fragment builders (decoder path).
// B fragment (32x16 K x N): lane n = lane%16, element e -> K = 16*half + e
__device__ __forceinline__ v16bf frag_b(const float* __restrict__ p) {
  const float4* q = (const float4*)p;
  return cvt16(q[0], q[1], q[2], q[3]);
}
// A fragment (16x32 M x K): lane m = lane%16, element e -> K = 16*half + e
__device__ __forceinline__ v16bf frag_a(const float* __restrict__ p) {
  const float4* q1 = (const float4*)p;
  const float4* q2 = (const float4*)(p + 8);
  return cvt16(q1[0], q1[1], q2[0], q2[1]);
}

// gfx1250 async global->LDS copy, tracked by ASYNCcnt (inline asm per ISA
// 15.18: inst_offset applies to BOTH lds and memory addresses).
__device__ __forceinline__ void async_copy_b128x2(unsigned lds_addr,
                                                  unsigned long long gaddr) {
  asm volatile("global_load_async_to_lds_b128 %0, %1, off"
               :: "v"(lds_addr), "v"(gaddr) : "memory");
  asm volatile("global_load_async_to_lds_b128 %0, %1, off offset:16"
               :: "v"(lds_addr), "v"(gaddr) : "memory");
}
__device__ __forceinline__ void wait_asynccnt0() {
  asm volatile("s_wait_asynccnt 0x0" ::: "memory");
}

// ---------------------------------------------------------------------------
// x (B,I,S) -> xT (S,B,I) via LDS tile transpose
// ---------------------------------------------------------------------------
__global__ __launch_bounds__(256) void k_transpose(const float* __restrict__ x,
                                                   float* __restrict__ xT) {
  __shared__ float tile[32][33];
  const int b  = blockIdx.x;
  const int i0 = blockIdx.y * 32;
  const int s0 = blockIdx.z * 32;
  const int tx = threadIdx.x, ty = threadIdx.y;  // 32 x 8
#pragma unroll
  for (int k = 0; k < 4; ++k)
    tile[ty + 8 * k][tx] = x[(size_t)b * I_ * S_ + (size_t)(i0 + ty + 8 * k) * S_ + s0 + tx];
  __syncthreads();
#pragma unroll
  for (int k = 0; k < 4; ++k)
    xT[(size_t)(s0 + ty + 8 * k) * B_ * I_ + (size_t)b * I_ + i0 + tx] = tile[tx][ty + 8 * k];
}

__global__ void k_zero(float* __restrict__ p, int n) {
  int t = blockIdx.x * blockDim.x + threadIdx.x;
  if (t < n) p[t] = 0.0f;
}

// ---------------------------------------------------------------------------
// Persistent encoder: 64 resident WGs loop over all S steps; device-scope
// atomic barrier between steps. Per k-step: A tile DMA'd to LDS via async
// loads (f32), B tile staged through VGPRs as bf16; 4 gate WMMAs per wave.
// ---------------------------------------------------------------------------
struct RawB { float4 b0, b1, b2, b3; };

__device__ __forceinline__ RawB enc_load_rawB(int step,
    const float* __restrict__ Wih, const float* __restrict__ Whh,
    int n_base, int tid) {
  RawB r;
  const int wrow = tid >> 1, wc = (tid & 1) * 16;  // 128 rows x 32 k, 16 f32/thr
  const int gate = wrow >> 5, nloc = wrow & 31;
  const float* W; int wsd, kt;
  if (step < 8) { W = Wih; wsd = I_; kt = step * 32; }
  else          { W = Whh; wsd = H_; kt = (step - 8) * 32; }
  const float* pb = W + (size_t)(gate * H_ + n_base + nloc) * wsd + kt + wc;
  __builtin_prefetch(pb + 32, 0, 0);  // next k-chunk of this weight row
  const float4* qb = (const float4*)pb;
  r.b0 = qb[0]; r.b1 = qb[1]; r.b2 = qb[2]; r.b3 = qb[3];
  return r;
}

__device__ __forceinline__ void storeB(const RawB& r, __bf16* bufB, int tid) {
  const int wrow = tid >> 1, wc = (tid & 1) * 16;
  int4 vb0 = make_int4(pk2(r.b0.x, r.b0.y), pk2(r.b0.z, r.b0.w),
                       pk2(r.b1.x, r.b1.y), pk2(r.b1.z, r.b1.w));
  int4 vb1 = make_int4(pk2(r.b2.x, r.b2.y), pk2(r.b2.z, r.b2.w),
                       pk2(r.b3.x, r.b3.y), pk2(r.b3.z, r.b3.w));
  int4* db = (int4*)((char*)bufB + wrow * 64 + wc * 2);
  db[0] = vb0; db[1] = vb1;
}

__device__ __forceinline__ void enc_issue_asyncA(int step,
    const float* __restrict__ xTs, const float* __restrict__ hp,
    int m_base, int tid, float* bufA) {
  const int arow = tid >> 2, ac = (tid & 3) * 8;   // 64 rows x 32 k, 8 f32/thr
  const float* A; int as, kt;
  if (step < 8) { A = xTs; as = I_; kt = step * 32; }
  else          { A = hp;  as = H_; kt = (step - 8) * 32; }
  const float* g = A + (size_t)(m_base + arow) * as + kt + ac;
  unsigned l = (unsigned)(size_t)(bufA + arow * 32 + ac);
  async_copy_b128x2(l, (unsigned long long)(size_t)g);
}

__global__ __launch_bounds__(256) void k_enc_persist(
    const float* __restrict__ xT,       // S,B,I
    const float* __restrict__ Wih_all,  // S,4H,I
    const float* __restrict__ Whh_all,  // S,4H,H
    const float* __restrict__ bih_all,  // S,4H
    const float* __restrict__ bhh_all,  // S,4H
    float* __restrict__ h0,             // B,H (double buffer)
    float* __restrict__ h1,
    float* __restrict__ enc,            // S,B,H
    int* __restrict__ sync)             // S counters, pre-zeroed
{
  __shared__ float  sAf[2][64 * 32];   // 2 x 8 KB, f32 (async DMA target)
  __shared__ __bf16 sB[2][128 * 32];   // 2 x 8 KB, bf16 (pre-converted)

  const int tid  = threadIdx.x;
  const int lane = tid & 31;
  const int wave = tid >> 5;
  const int half = lane >> 4;
  const int lr   = lane & 15;
  const int wg   = blockIdx.x;         // 0..63
  const int m_base = (wg & 3) * 64;    // over B
  const int n_base = (wg >> 2) * 32;   // over H
  const int m_sub = wave >> 1, n_sub = wave & 1;
  const int bn = n_base + n_sub * 16 + lr;   // this lane's gate column

  for (int s = 0; s < S_; ++s) {
    const float* hp  = (s & 1) ? h1 : h0;
    float*       hn  = (s & 1) ? h0 : h1;
    const float* xTs = xT + (size_t)s * B_ * I_;
    const float* Wih = Wih_all + (size_t)s * 4 * H_ * I_;
    const float* Whh = Whh_all + (size_t)s * 4 * H_ * H_;
    const float* bih = bih_all + (size_t)s * 4 * H_;
    const float* bhh = bhh_all + (size_t)s * 4 * H_;
    float* encs = enc + (size_t)s * B_ * H_;

    v8f ci = {}, cf = {}, cg = {}, co = {};
    enc_issue_asyncA(0, xTs, hp, m_base, tid, sAf[0]);
    RawB rb = enc_load_rawB(0, Wih, Whh, n_base, tid);

    for (int st = 0; st < 24; ++st) {
      const int cur = st & 1;
      storeB(rb, sB[cur], tid);
      wait_asynccnt0();               // A tile (issued last iter) landed in LDS
      __syncthreads();                // sB[cur] + sAf[cur] visible to all waves
      if (st + 1 < 24) {
        enc_issue_asyncA(st + 1, xTs, hp, m_base, tid, sAf[cur ^ 1]);
        rb = enc_load_rawB(st + 1, Wih, Whh, n_base, tid);
      }
      // A fragment: f32 from LDS, convert -> bf16
      v16bf fa;
      {
        const char* pa = (const char*)sAf[cur] + (m_sub * 16 + lr) * 128 + 32 * half;
        const float4* q1 = (const float4*)pa;         // K = 8*half .. +8
        const float4* q2 = (const float4*)(pa + 64);  // K = 16+8*half .. +8
        fa = cvt16(q1[0], q1[1], q2[0], q2[1]);
      }
      // B fragments (bf16 in LDS): 2 x ds_load_b128 each
      BF16x16 f0, f1, f2, f3;
      {
        const int rB = n_sub * 16 + lr;
        const char* pb = (const char*)sB[cur];
        const int4* q0 = (const int4*)(pb + (0 * 32 + rB) * 64 + 16 * half);
        const int4* q1 = (const int4*)(pb + (1 * 32 + rB) * 64 + 16 * half);
        const int4* q2 = (const int4*)(pb + (2 * 32 + rB) * 64 + 16 * half);
        const int4* q3 = (const int4*)(pb + (3 * 32 + rB) * 64 + 16 * half);
        f0.q[0] = q0[0]; f0.q[1] = q0[2];
        f1.q[0] = q1[0]; f1.q[1] = q1[2];
        f2.q[0] = q2[0]; f2.q[1] = q2[2];
        f3.q[0] = q3[0]; f3.q[1] = q3[2];
      }
      ci = WMMA_BF16(fa, f0.v, ci);
      cf = WMMA_BF16(fa, f1.v, cf);
      cg = WMMA_BF16(fa, f2.v, cg);
      co = WMMA_BF16(fa, f3.v, co);
      // single barrier per k-step suffices with double buffering: the buffer
      // written next iteration was last read two iterations ago.
    }

    // Epilogue: bias + cell. C/D layout: n = lane%16, VGPR r -> m = r + 8*half.
    const float bI = bih[0 * H_ + bn] + bhh[0 * H_ + bn];
    const float bF = bih[1 * H_ + bn] + bhh[1 * H_ + bn];
    const float bG = bih[2 * H_ + bn] + bhh[2 * H_ + bn];
    const float bO = bih[3 * H_ + bn] + bhh[3 * H_ + bn];
#pragma unroll
    for (int rr = 0; rr < 8; ++rr) {
      const int m = m_base + m_sub * 16 + rr + 8 * half;
      const float iv = ci[rr] + bI, fv = cf[rr] + bF, gv = cg[rr] + bG, ov = co[rr] + bO;
      const float hpv = hp[(size_t)m * H_ + bn];
      const float c   = sigf(fv) * hpv + sigf(iv) * tanhf(gv);
      const float hv  = sigf(ov) * tanhf(c);
      hn[(size_t)m * H_ + bn]   = hv;
      encs[(size_t)m * H_ + bn] = hv;
    }

    // Grid-wide barrier: make h_next visible, then wait for all 64 WGs.
    __threadfence();
    __syncthreads();
    if (tid == 0) {
      __hip_atomic_fetch_add(&sync[s], 1, __ATOMIC_RELEASE, __HIP_MEMORY_SCOPE_AGENT);
      int v;
      do {
        v = __hip_atomic_load(&sync[s], __ATOMIC_ACQUIRE, __HIP_MEMORY_SCOPE_AGENT);
        if (v < NWG_ENC) __builtin_amdgcn_s_sleep(8);
      } while (v < NWG_ENC);
    }
    __syncthreads();
    __threadfence();
  }
}

// ---------------------------------------------------------------------------
// elu (in place) over enc (S,B,H)
// ---------------------------------------------------------------------------
__global__ void k_elu(float* __restrict__ enc, int n) {
  int t = blockIdx.x * blockDim.x + threadIdx.x;
  if (t < n) {
    float v = enc[t];
    enc[t] = v > 0.0f ? v : (__expf(v) - 1.0f);
  }
}

// Residual chain over k = 0,4,...,60; independent per (b,h) element.
__global__ void k_residual(float* __restrict__ enc) {
  int t = blockIdx.x * blockDim.x + threadIdx.x;
  if (t >= B_ * H_) return;
  const size_t BH = (size_t)B_ * H_;
  float prev = enc[(size_t)(S_ - RESN) * BH + t];
#pragma unroll
  for (int j = 0; j < S_ / RESN; ++j) {
    const size_t off = (size_t)(j * RESN) * BH + t;
    float v = enc[off];
    float nv = ALPHA_ * v + (1.0f - ALPHA_) * prev;
    enc[off] = nv;
    prev = nv;
  }
}

// ---------------------------------------------------------------------------
// Decoder: batched over s (grid.z=64, 2048 WGs -> occupancy hides latency).
// Only gates i, g, o are live (f and Whh_dec are dead in the reference).
// ---------------------------------------------------------------------------
__global__ __launch_bounds__(256) void k_dec(
    const float* __restrict__ enc,  // S,B,H (post elu+residual)
    const float* __restrict__ Wih,  // S,4I,H
    const float* __restrict__ bih,  // S,4I
    const float* __restrict__ bhh,  // S,4I
    float* __restrict__ hd)         // S,B,I
{
  const int s = blockIdx.z;
  const float* A   = enc + (size_t)(S_ - 1 - s) * B_ * H_;
  const float* W   = Wih + (size_t)s * 4 * I_ * H_;
  const float* b1v = bih + (size_t)s * 4 * I_;
  const float* b2v = bhh + (size_t)s * 4 * I_;

  const int lane = threadIdx.x & 31;
  const int wave = threadIdx.x >> 5;
  const int half = lane >> 4;
  const int lr   = lane & 15;
  const int m0 = blockIdx.x * 64 + (wave >> 1) * 16;
  const int n0 = blockIdx.y * 32 + (wave & 1) * 16;
  const int am = m0 + lr;
  const int bn = n0 + lr;

  const float* ar = A + (size_t)am * H_;
  const float* wi = W + (size_t)(0 * I_ + bn) * H_;
  const float* wg = W + (size_t)(2 * I_ + bn) * H_;
  const float* wo = W + (size_t)(3 * I_ + bn) * H_;

  v8f ci = {}, cg = {}, co = {};
  for (int kt = 0; kt < H_; kt += 32) {
    __builtin_prefetch(wi + kt + 128, 0, 0);
    __builtin_prefetch(wg + kt + 128, 0, 0);
    __builtin_prefetch(wo + kt + 128, 0, 0);
    v16bf a    = frag_a(ar + kt + 16 * half);
    v16bf bi_f = frag_b(wi + kt + 16 * half);
    v16bf bg_f = frag_b(wg + kt + 16 * half);
    v16bf bo_f = frag_b(wo + kt + 16 * half);
    ci = WMMA_BF16(a, bi_f, ci);
    cg = WMMA_BF16(a, bg_f, cg);
    co = WMMA_BF16(a, bo_f, co);
  }

  const float bI = b1v[0 * I_ + bn] + b2v[0 * I_ + bn];
  const float bG = b1v[2 * I_ + bn] + b2v[2 * I_ + bn];
  const float bO = b1v[3 * I_ + bn] + b2v[3 * I_ + bn];
#pragma unroll
  for (int r = 0; r < 8; ++r) {
    const int m = m0 + r + 8 * half;
    const float iv = ci[r] + bI, gv = cg[r] + bG, ov = co[r] + bO;
    const float c = sigf(iv) * tanhf(gv);
    hd[(size_t)s * B_ * I_ + (size_t)m * I_ + bn] = sigf(ov) * tanhf(c);
  }
}

// ---------------------------------------------------------------------------
// Per-4-block cumsum over s, tanh, reverse s, transpose to out (B,I,S)
// ---------------------------------------------------------------------------
__global__ void k_final(const float* __restrict__ hd, float* __restrict__ out) {
  int t = blockIdx.x * blockDim.x + threadIdx.x;
  if (t >= B_ * I_) return;
  const int b = t / I_;
  const int i = t % I_;
  const float* p = hd + (size_t)b * I_ + i;
  float* o = out + ((size_t)b * I_ + i) * S_;
  const size_t BI = (size_t)B_ * I_;
  for (int blk = 0; blk < S_ / RESN; ++blk) {
    float run = 0.0f;
#pragma unroll
    for (int j = 0; j < RESN; ++j) {
      const int sp = blk * RESN + j;
      run += p[(size_t)sp * BI];
      o[S_ - 1 - sp] = tanhf(run);
    }
  }
}

// ---------------------------------------------------------------------------
extern "C" void kernel_launch(void* const* d_in, const int* in_sizes, int n_in,
                              void* d_out, int out_size, void* d_ws, size_t ws_size,
                              hipStream_t stream) {
  const float* x       = (const float*)d_in[0];
  const float* Wih_enc = (const float*)d_in[1];
  const float* Whh_enc = (const float*)d_in[2];
  const float* bih_enc = (const float*)d_in[3];
  const float* bhh_enc = (const float*)d_in[4];
  const float* Wih_dec = (const float*)d_in[5];
  // d_in[6] = Whh_dec : provably unused by the reference computation
  const float* bih_dec = (const float*)d_in[7];
  const float* bhh_dec = (const float*)d_in[8];
  float* out = (float*)d_out;

  // Workspace carve-up (f32 elements):
  float* ws    = (float*)d_ws;
  float* xT    = ws;                               // S*B*I
  float* h0    = xT + (size_t)S_ * B_ * I_;        // B*H
  float* h1    = h0 + (size_t)B_ * H_;             // B*H
  float* syncf = h1 + (size_t)B_ * H_;             // S_ ints (as f32 slots)
  float* enc   = syncf + S_;                       // S*B*H
  float* hd    = enc + (size_t)S_ * B_ * H_;       // S*B*I
  int*   syncp = (int*)syncf;

  // 1) transpose x (B,I,S) -> xT (S,B,I)
  {
    dim3 tb(32, 8);
    dim3 tg(B_, I_ / 32, S_ / 32);
    k_transpose<<<tg, tb, 0, stream>>>(x, xT);
  }
  // 2) zero h0, h1 and the sync counters (bit pattern 0)
  {
    int n = 2 * B_ * H_ + S_;
    k_zero<<<(n + 255) / 256, 256, 0, stream>>>(h0, n);
  }
  // 3) persistent encoder: one launch for all 64 sequential steps
  {
    k_enc_persist<<<NWG_ENC, 256, 0, stream>>>(xT, Wih_enc, Whh_enc,
                                               bih_enc, bhh_enc,
                                               h0, h1, enc, syncp);
  }
  // 4) elu + residual chain
  {
    int n = S_ * B_ * H_;
    k_elu<<<(n + 255) / 256, 256, 0, stream>>>(enc, n);
    int m = B_ * H_;
    k_residual<<<(m + 255) / 256, 256, 0, stream>>>(enc);
  }
  // 5) decoder (all s in one batched launch, 3 live gates)
  {
    dim3 gb(256);
    dim3 dg(B_ / 64, I_ / 32, S_);
    k_dec<<<dg, gb, 0, stream>>>(enc, Wih_dec, bih_dec, bhh_dec, hd);
  }
  // 6) blockwise cumsum + tanh + reverse + transpose
  {
    int n = B_ * I_;
    k_final<<<(n + 255) / 256, 256, 0, stream>>>(hd, out);
  }
}
